// SpikingSelfAttention_21655225106977
// MI455X (gfx1250) — 16-bit and 8-bit matrix operand order repaired and checked by reading, compiled, not run on hardware
//
#include <hip/hip_runtime.h>
#include <stdint.h>

#define TT  4
#define BB  4
#define CC  512
#define NNp 1024
#define NHH 8
#define HDD 64

typedef __attribute__((ext_vector_type(16))) _Float16 v16h;
typedef __attribute__((ext_vector_type(8)))  _Float16 v8h;
typedef __attribute__((ext_vector_type(8)))  float    v8f;
typedef __attribute__((ext_vector_type(8)))  int      v8i;

__global__ void prep_w(const float* __restrict__ W, _Float16* __restrict__ W16, int n) {
    int i = blockIdx.x * blockDim.x + threadIdx.x;
    if (i < n) W16[i] = (_Float16)W[i];
}

__global__ void prep_bn(const float* __restrict__ g, const float* __restrict__ b,
                        const float* __restrict__ m, const float* __restrict__ v,
                        const float* __restrict__ bias,
                        float* __restrict__ scale, float* __restrict__ shift) {
    int i = blockIdx.x * blockDim.x + threadIdx.x;
    if (i < CC) {
        float sc = g[i] * rsqrtf(v[i] + 1e-5f);
        float sh = b[i] - m[i] * sc;
        if (bias) sh += bias[i] * sc;
        scale[i] = sc;
        shift[i] = sh;
    }
}

__global__ void lif_in(const float* __restrict__ x, _Float16* __restrict__ xs) {
    const int S = BB * CC * NNp;
    int i = blockIdx.x * blockDim.x + threadIdx.x;
    if (i >= S) return;
    float vm = 0.f;
#pragma unroll
    for (int t = 0; t < TT; ++t) {
        float xt = x[(size_t)t * S + i];
        vm += (xt - vm) * 0.5f;
        float s = (vm >= 1.0f) ? 1.f : 0.f;
        xs[(size_t)t * S + i] = (_Float16)s;
        if (s > 0.f) vm = 0.f;
    }
}

#define BPAD 24
__global__ __launch_bounds__(256)
void qkv_gemm(const _Float16* __restrict__ W16, const _Float16* __restrict__ xs,
              const float* __restrict__ scale, const float* __restrict__ shift,
              uint8_t* __restrict__ out, int kLayout) {
    __shared__ alignas(16) _Float16 bstage[2][TT][32][BPAD];

    const int tid  = threadIdx.x;
    const int lane = tid & 31;
    const int wave = tid >> 5;
    const int n0   = blockIdx.x * 16;
    const int d0   = blockIdx.y * 128 + wave * 16;
    const int b    = blockIdx.z;
    const int half = lane >> 4;
    const int l16  = lane & 15;

    const int ct   = tid >> 6;
    const int crow = (tid >> 1) & 31;
    const int cnh  = tid & 1;
    const uint64_t gsrc0 =
        (uint64_t)(uintptr_t)(xs + ((size_t)(ct*BB + b) * CC + crow) * NNp + n0 + cnh * 8);
    const uint32_t ldst0 = (uint32_t)(uintptr_t)&bstage[0][ct][crow][cnh * 8];
    const uint32_t ldst1 = (uint32_t)(uintptr_t)&bstage[1][ct][crow][cnh * 8];

    const _Float16* Wrow = W16 + (size_t)(d0 + l16) * CC + half * 8;

    v8f acc[TT] = {};

    auto step = [&](auto bufc, int cb, int cbn, uint32_t ldNext) {
        constexpr int BUFI = decltype(bufc)::value;
        asm volatile("global_load_async_to_lds_b128 %0, %1, off"
                     :: "v"(ldNext), "v"(gsrc0 + (uint64_t)cbn * (NNp * 2)) : "memory");
        asm volatile("s_wait_asynccnt 0x1" ::: "memory");
        __syncthreads();

        v8h a0 = *(const v8h*)(Wrow + cb);
        v8h a1 = *(const v8h*)(Wrow + cb + 16);
        v16h A = __builtin_shufflevector(a0, a1, 0,1,2,3,4,5,6,7,8,9,10,11,12,13,14,15);

        v16h Bv[TT];
#pragma unroll
        for (int t = 0; t < TT; ++t) {
            const _Float16* br = &bstage[BUFI][t][half * 8][l16];
#pragma unroll
            for (int s = 0; s < 8; ++s) {
                Bv[t][s]     = br[s * BPAD];
                Bv[t][s + 8] = br[(16 + s) * BPAD];
            }
        }
#pragma unroll
        for (int t = 0; t < TT; ++t)
            acc[t] = __builtin_amdgcn_wmma_f32_16x16x32_f16(
                false, A, false, Bv[t], (short)0, acc[t], false, false);

        __syncthreads();
    };

    asm volatile("global_load_async_to_lds_b128 %0, %1, off"
                 :: "v"(ldst0), "v"(gsrc0) : "memory");

    for (int ii = 0; ii < 8; ++ii) {
        const int cb = ii * 64;
        step(std::integral_constant<int,0>{}, cb,      cb + 32,                    ldst1);
        step(std::integral_constant<int,1>{}, cb + 32, (ii < 7) ? cb + 64 : 0,     ldst0);
    }

    const int dd = d0 + half * 8;
    float4 s0 = *(const float4*)(scale + dd);
    float4 s1 = *(const float4*)(scale + dd + 4);
    float4 h0 = *(const float4*)(shift + dd);
    float4 h1 = *(const float4*)(shift + dd + 4);
    float scv[8] = {s0.x, s0.y, s0.z, s0.w, s1.x, s1.y, s1.z, s1.w};
    float shv[8] = {h0.x, h0.y, h0.z, h0.w, h1.x, h1.y, h1.z, h1.w};

    const int h   = d0 >> 6;
    const int hd0 = (d0 & 63) + half * 8;
    const int n   = n0 + l16;

    float vmem[8] = {0.f,0.f,0.f,0.f,0.f,0.f,0.f,0.f};
#pragma unroll
    for (int t = 0; t < TT; ++t) {
        uint64_t pk = 0;
#pragma unroll
        for (int r = 0; r < 8; ++r) {
            float y = acc[t][r] * scv[r] + shv[r];
            vmem[r] += (y - vmem[r]) * 0.5f;
            uint32_t s = (vmem[r] >= 1.0f) ? 1u : 0u;
            pk |= (uint64_t)s << (8 * r);
            if (s) vmem[r] = 0.f;
        }
        if (kLayout == 0) {
            uint8_t* p = out + (((size_t)(t*BB + b) * NHH + h) * NNp + n) * HDD + hd0;
            *(uint64_t*)p = pk;
        } else {
            uint8_t* p = out + (((size_t)(t*BB + b) * NHH + h) * HDD + hd0) * NNp + n;
#pragma unroll
            for (int r = 0; r < 8; ++r) p[(size_t)r * NNp] = (uint8_t)(pk >> (8 * r));
        }
    }
}

__global__ __launch_bounds__(128)
void attention(const uint8_t* __restrict__ q, const uint8_t* __restrict__ kt,
               const uint8_t* __restrict__ v, _Float16* __restrict__ os) {
    __shared__ alignas(16) uint8_t attn_s[TT][16][64];
    __shared__ float o_s[TT][16 * 64];

    const int lane = threadIdx.x & 31;
    const int t    = threadIdx.x >> 5;
    const int n0   = blockIdx.x * 16;
    const int h    = blockIdx.y;
    const int b    = blockIdx.z;
    const int half = lane >> 4;
    const int l16  = lane & 15;

    const size_t head  = (size_t)(t*BB + b) * NHH + h;
    const uint8_t* qrow  = q  + (head * NNp + n0 + l16) * HDD + half * 8;
    const uint8_t* kbase = kt + head * HDD * NNp;
    const uint8_t* vbase = v  + head * NNp * HDD;

    v8i Aq;
    {
        uint2 c0 = *(const uint2*)(qrow +  0);
        uint2 c1 = *(const uint2*)(qrow + 16);
        uint2 c2 = *(const uint2*)(qrow + 32);
        uint2 c3 = *(const uint2*)(qrow + 48);
        Aq[0]=(int)c0.x; Aq[1]=(int)c0.y; Aq[2]=(int)c1.x; Aq[3]=(int)c1.y;
        Aq[4]=(int)c2.x; Aq[5]=(int)c2.y; Aq[6]=(int)c3.x; Aq[7]=(int)c3.y;
    }

    v8i oacc[4] = {};
    const v8i zero = {};

    for (int mb = 0; mb < NNp; mb += 64) {
        if (mb + 64 < NNp) {
            __builtin_prefetch(kbase + (size_t)lane * NNp + mb + 64, 0, 0);
            __builtin_prefetch(vbase + (size_t)(mb + 64 + lane) * HDD, 0, 0);
        }
#pragma unroll
        for (int sub = 0; sub < 4; ++sub) {
            const int m0 = mb + sub * 16;
            const uint8_t* kc = kbase + m0 + l16;
            v8i Bk;
#pragma unroll
            for (int j = 0; j < 8; ++j) {
                uint32_t w = 0;
#pragma unroll
                for (int e = 0; e < 4; ++e) {
                    const int s = j * 4 + e;
                    const int kk = 16 * (s >> 3) + half * 8 + (s & 7);
                    w |= (uint32_t)kc[(size_t)kk * NNp] << (8 * e);
                }
                Bk[j] = (int)w;
            }
            v8i at = __builtin_amdgcn_wmma_i32_16x16x64_iu8(
                false, Aq, false, Bk, zero, false, false);
#pragma unroll
            for (int r = 0; r < 8; ++r)
                attn_s[t][r + half * 8][sub * 16 + l16] = (uint8_t)at[r];
        }
        v8i Aa;
        {
            const uint8_t* ar = &attn_s[t][l16][half * 8];
            uint2 c0 = *(const uint2*)(ar +  0);
            uint2 c1 = *(const uint2*)(ar + 16);
            uint2 c2 = *(const uint2*)(ar + 32);
            uint2 c3 = *(const uint2*)(ar + 48);
            Aa[0]=(int)c0.x; Aa[1]=(int)c0.y; Aa[2]=(int)c1.x; Aa[3]=(int)c1.y;
            Aa[4]=(int)c2.x; Aa[5]=(int)c2.y; Aa[6]=(int)c3.x; Aa[7]=(int)c3.y;
        }
#pragma unroll
        for (int hs = 0; hs < 4; ++hs) {
            const int hd0 = hs * 16;
            const uint8_t* vc = vbase + (size_t)mb * HDD + hd0 + l16;
            v8i Bv;
#pragma unroll
            for (int j = 0; j < 8; ++j) {
                uint32_t w = 0;
#pragma unroll
                for (int e = 0; e < 4; ++e) {
                    const int s = j * 4 + e;
                    const int kk = 16 * (s >> 3) + half * 8 + (s & 7);
                    w |= (uint32_t)vc[(size_t)kk * HDD] << (8 * e);
                }
                Bv[j] = (int)w;
            }
            oacc[hs] = __builtin_amdgcn_wmma_i32_16x16x64_iu8(
                false, Aa, false, Bv, oacc[hs], false, false);
        }
    }

#pragma unroll
    for (int hs = 0; hs < 4; ++hs)
#pragma unroll
        for (int r = 0; r < 8; ++r)
            o_s[t][(r + half * 8) * 64 + hs * 16 + l16] = (float)oacc[hs][r] * 0.125f;

    __syncthreads();

    const int tid = threadIdx.x;
#pragma unroll
    for (int j = 0; j < 8; ++j) {
        const int e  = tid * 8 + j;
        const int nl = e >> 6;
        const int hd = e & 63;
        float vm = 0.f;
#pragma unroll
        for (int t2 = 0; t2 < TT; ++t2) {
            float y = o_s[t2][e];
            vm += (y - vm) * 0.5f;
            float s = (vm >= 0.5f) ? 1.f : 0.f;
            os[((size_t)(t2*BB + b) * CC + h * HDD + hd) * NNp + n0 + nl] = (_Float16)s;
            if (s > 0.f) vm = 0.f;
        }
    }
}

__global__ __launch_bounds__(256)
void proj(const _Float16* __restrict__ Wp16, const _Float16* __restrict__ os,
          const float* __restrict__ scale, const float* __restrict__ shift,
          float* __restrict__ out) {
    const int lane = threadIdx.x & 31;
    const int wave = threadIdx.x >> 5;
    const int n0   = blockIdx.x * 16;
    const int d0   = blockIdx.y * 128 + wave * 16;
    const int tb   = blockIdx.z;
    const int half = lane >> 4;
    const int l16  = lane & 15;

    const _Float16* Wrow  = Wp16 + (size_t)(d0 + l16) * CC + half * 8;
    const _Float16* xbase = os + (size_t)tb * CC * NNp;

    v8f acc = {};
    for (int cb = 0; cb < CC; cb += 32) {
        v8h a0 = *(const v8h*)(Wrow + cb);
        v8h a1 = *(const v8h*)(Wrow + cb + 16);
        v16h A = __builtin_shufflevector(a0, a1, 0,1,2,3,4,5,6,7,8,9,10,11,12,13,14,15);
        const _Float16* xr = xbase + (size_t)(cb + half * 8) * NNp + n0 + l16;
        v16h Bv;
#pragma unroll
        for (int s = 0; s < 8; ++s) {
            Bv[s]     = xr[(size_t)s * NNp];
            Bv[s + 8] = xr[(size_t)(16 + s) * NNp];
        }
        acc = __builtin_amdgcn_wmma_f32_16x16x32_f16(
            false, A, false, Bv, (short)0, acc, false, false);
    }

    const int dd = d0 + half * 8;
    float4 s0 = *(const float4*)(scale + dd);
    float4 s1 = *(const float4*)(scale + dd + 4);
    float4 h0 = *(const float4*)(shift + dd);
    float4 h1 = *(const float4*)(shift + dd + 4);
    float scv[8] = {s0.x, s0.y, s0.z, s0.w, s1.x, s1.y, s1.z, s1.w};
    float shv[8] = {h0.x, h0.y, h0.z, h0.w, h1.x, h1.y, h1.z, h1.w};

    float* obase = out + ((size_t)tb * CC + dd) * NNp + n0 + l16;
#pragma unroll
    for (int r = 0; r < 8; ++r)
        obase[(size_t)r * NNp] = acc[r] * scv[r] + shv[r];
}

extern "C" void kernel_launch(void* const* d_in, const int* in_sizes, int n_in,
                              void* d_out, int out_size, void* d_ws, size_t ws_size,
                              hipStream_t stream) {
    const float* x   = (const float*)d_in[0];
    const float* Wq  = (const float*)d_in[1];
    const float* q_g = (const float*)d_in[2];
    const float* q_b = (const float*)d_in[3];
    const float* q_m = (const float*)d_in[4];
    const float* q_v = (const float*)d_in[5];
    const float* Wk  = (const float*)d_in[6];
    const float* k_g = (const float*)d_in[7];
    const float* k_b = (const float*)d_in[8];
    const float* k_m = (const float*)d_in[9];
    const float* k_v = (const float*)d_in[10];
    const float* Wv  = (const float*)d_in[11];
    const float* v_g = (const float*)d_in[12];
    const float* v_b = (const float*)d_in[13];
    const float* v_m = (const float*)d_in[14];
    const float* v_v = (const float*)d_in[15];
    const float* Wp  = (const float*)d_in[16];
    const float* bp  = (const float*)d_in[17];
    const float* p_g = (const float*)d_in[18];
    const float* p_b = (const float*)d_in[19];
    const float* p_m = (const float*)d_in[20];
    const float* p_v = (const float*)d_in[21];
    (void)in_sizes; (void)n_in; (void)out_size; (void)ws_size;

    uint8_t* ws8 = (uint8_t*)d_ws;
    _Float16* xs   = (_Float16*)ws8;
    uint8_t*  ktb  = ws8 + ((size_t)16 << 20);
    _Float16* W16  = (_Float16*)(ws8 + ((size_t)24 << 20));
    _Float16* Wq16 = W16;
    _Float16* Wk16 = W16 + 262144;
    _Float16* Wv16 = W16 + 2 * 262144;
    _Float16* Wp16 = W16 + 3 * 262144;
    float* bn  = (float*)(ws8 + ((size_t)26 << 20));
    float *qsc = bn,        *qsh = bn + 512;
    float *ksc = bn + 1024, *ksh = bn + 1536;
    float *vsc = bn + 2048, *vsh = bn + 2560;
    float *psc = bn + 3072, *psh = bn + 3584;

    uint8_t* qbuf = (uint8_t*)d_out;
    uint8_t* vbuf = (uint8_t*)d_out + ((size_t)8 << 20);

    prep_w<<<1024, 256, 0, stream>>>(Wq, Wq16, 262144);
    prep_w<<<1024, 256, 0, stream>>>(Wk, Wk16, 262144);
    prep_w<<<1024, 256, 0, stream>>>(Wv, Wv16, 262144);
    prep_w<<<1024, 256, 0, stream>>>(Wp, Wp16, 262144);
    prep_bn<<<2, 256, 0, stream>>>(q_g, q_b, q_m, q_v, nullptr, qsc, qsh);
    prep_bn<<<2, 256, 0, stream>>>(k_g, k_b, k_m, k_v, nullptr, ksc, ksh);
    prep_bn<<<2, 256, 0, stream>>>(v_g, v_b, v_m, v_v, nullptr, vsc, vsh);
    prep_bn<<<2, 256, 0, stream>>>(p_g, p_b, p_m, p_v, bp,      psc, psh);

    lif_in<<<(BB * CC * NNp) / 256, 256, 0, stream>>>(x, xs);

    dim3 gG(NNp / 16, CC / 128, BB);
    qkv_gemm<<<gG, 256, 0, stream>>>(Wq16, xs, qsc, qsh, qbuf, 0);
    qkv_gemm<<<gG, 256, 0, stream>>>(Wk16, xs, ksc, ksh, ktb, 1);
    qkv_gemm<<<gG, 256, 0, stream>>>(Wv16, xs, vsc, vsh, vbuf, 0);

    dim3 gA(NNp / 16, NHH, BB);
    attention<<<gA, 128, 0, stream>>>(qbuf, ktb, vbuf, xs);

    dim3 gP(NNp / 16, CC / 128, TT * BB);
    proj<<<gP, 256, 0, stream>>>(Wp16, xs, psc, psh, (float*)d_out);
}
